// HSConv_90924457656405
// MI455X (gfx1250) — compile-verified, matrix operand order checked by reading, not run on hardware
//
#include <hip/hip_runtime.h>

// ---------------------------------------------------------------------------
// CDNA5 (gfx1250) HSConv GNN layer.
//
// Roofline: ~22.6 GFLOP vs ~1.4 GB of HBM traffic -> memory bound at 23.3TB/s
// (~60us floor). All GEMMs run on v_wmma_f32_16x16x32_bf16 (confirmed CDNA5
// builtin). Weights (40K fp32 total) are packed once per launch into bf16 in
// the exact WMMA B-operand lane layout; feature tiles are staged in LDS and
// packed into the A-operand layout on the fly using the HARDWARE f32->bf16
// converter (plain fptrunc cast -> v_cvt*bf16* instructions), not a software
// bit-twiddling sequence.
//
// WMMA 16x16x32 bf16 operand layouts used (CDNA5 ISA 7.12.2):
//   A (16xK=32): lane l -> row m = l&15, half = l>>4;
//                elems 0..7  = K (kb + half*8 .. +7)
//                elems 8..15 = K (kb + 16 + half*8 .. +7)
//   B (K=32x16): symmetric with A (lane indexes N, elems index K).
//   C/D (16x16 f32, 8 VGPRs): lane l -> col n = l&15; vgpr r -> row
//                m = (l>=16 ? 8 : 0) + r.
// ---------------------------------------------------------------------------

typedef __attribute__((ext_vector_type(16))) __bf16 v16bf;
typedef __attribute__((ext_vector_type(8)))  float  v8f;

// Hardware conversion: let the backend select v_cvt_(pk_)bf16_f32.
__device__ __forceinline__ __bf16 cvt_bf16(float f) {
  return (__bf16)f;
}

__device__ __forceinline__ v8f wmma_bf16(v16bf a, v16bf b, v8f c) {
  return __builtin_amdgcn_wmma_f32_16x16x32_bf16(
      /*neg_a=*/false, a, /*neg_b=*/false, b,
      /*c_mod=*/(short)0, c, /*reuse_a=*/false, /*reuse_b=*/false);
}

// Pack one lane's A-operand (16 bf16 = 2 contiguous groups of 8 K-values)
// from a fp32 row (LDS), K-tile base kb, lane half (l>>4).
__device__ __forceinline__ v16bf pack_a(const float* row, int kb, int half) {
  v16bf a;
  const float* p0 = row + kb + half * 8;
  const float* p1 = p0 + 16;
#pragma unroll
  for (int j = 0; j < 8; ++j) {
    a[j]     = cvt_bf16(p0[j]);
    a[j + 8] = cvt_bf16(p1[j]);
  }
  return a;
}

// Pre-swizzled B tile: tile (kt,nt) is 512 bf16 = 32 lanes x 16 elems.
__device__ __forceinline__ v16bf load_b(const __bf16* p, int kt, int nt, int lane) {
  return *reinterpret_cast<const v16bf*>(p + (((kt << 2) + nt) << 9) + (lane << 4));
}

// ---------------------------------------------------------------------------
// Weight packer: fp32 W[K x 64] row-major -> bf16 WMMA-B tiles.
// out[idx] with idx = ((kt*4+nt)*32 + lane)*16 + i holds
// W[kt*32 + (i&8?16:0) + (lane>>4)*8 + (i&7)][nt*16 + (lane&15)].
// ---------------------------------------------------------------------------
__global__ void pack_weight_kernel(const float* __restrict__ W,
                                   __bf16* __restrict__ out, int K) {
  int idx = blockIdx.x * 256 + threadIdx.x;
  int total = K * 64;
  if (idx >= total) return;
  int tile = idx >> 9;
  int rem  = idx & 511;
  int lane = rem >> 4;
  int i    = rem & 15;
  int kt = tile >> 2, nt = tile & 3;
  int half = lane >> 4, n = lane & 15;
  int k = kt * 32 + ((i & 8) ? 16 : 0) + half * 8 + (i & 7);
  out[idx] = cvt_bf16(W[k * 64 + nt * 16 + n]);
}

// ---------------------------------------------------------------------------
// Node pre-GEMMs: h1 = X*Wu, hu = X*Weu, hv = [S,X]*Wev.
// One block (128 thr = 4 waves) per 16-node tile; wave w = column tile w.
// ---------------------------------------------------------------------------
__global__ void __launch_bounds__(128)
node_pre_kernel(const float* __restrict__ S, const float* __restrict__ X,
                const __bf16* __restrict__ pWu, const __bf16* __restrict__ pWeu,
                const __bf16* __restrict__ pWev,
                float* __restrict__ h1, float* __restrict__ hu,
                float* __restrict__ hv, int N) {
  __shared__ float s_lds[16 * 64];
  __shared__ float x_lds[16 * 64];
  const int row0 = blockIdx.x * 16;

  for (int idx = threadIdx.x; idx < 16 * 64; idx += 128) {
    int r = idx >> 6, c = idx & 63;
    int node = min(row0 + r, N - 1);
    s_lds[idx] = S[(size_t)node * 64 + c];
    x_lds[idx] = X[(size_t)node * 64 + c];
  }
  __syncthreads();

  const int lane = threadIdx.x & 31;
  const int nt   = threadIdx.x >> 5;
  const int half = lane >> 4;
  const float* xrow = x_lds + (lane & 15) * 64;
  const float* srow = s_lds + (lane & 15) * 64;

  v8f a1 = {}, a2 = {}, a3 = {};
#pragma unroll
  for (int kt = 0; kt < 2; ++kt) {
    v16bf ax = pack_a(xrow, kt * 32, half);
    a1 = wmma_bf16(ax, load_b(pWu,  kt, nt, lane), a1);
    a2 = wmma_bf16(ax, load_b(pWeu, kt, nt, lane), a2);
    v16bf as = pack_a(srow, kt * 32, half);          // hv K 0..63 = S
    a3 = wmma_bf16(as, load_b(pWev, kt, nt, lane), a3);
  }
#pragma unroll
  for (int kt = 2; kt < 4; ++kt) {                   // hv K 64..127 = X
    v16bf ax = pack_a(xrow, (kt - 2) * 32, half);
    a3 = wmma_bf16(ax, load_b(pWev, kt, nt, lane), a3);
  }

  const int col = nt * 16 + (lane & 15);
  const int mb  = half * 8;
#pragma unroll
  for (int r = 0; r < 8; ++r) {
    int node = row0 + mb + r;
    if (node < N) {
      size_t o = (size_t)node * 64 + col;
      h1[o] = a1[r];
      hu[o] = a2[r];
      hv[o] = a3[r];
    }
  }
}

// ---------------------------------------------------------------------------
// Edge kernel: e1 = Ein*We2n; m = h1[src]+e1 -> atomic segment-sum on dst;
// e_out = [Ein, hu[src]+hv[dst]]*We2e + bias_e. One block per 16-edge tile.
// ---------------------------------------------------------------------------
__global__ void __launch_bounds__(128)
edge_kernel(const float* __restrict__ Ein, const int* __restrict__ src,
            const int* __restrict__ dst, const float* __restrict__ h1,
            const float* __restrict__ hu, const float* __restrict__ hv,
            const __bf16* __restrict__ pWe2n, const __bf16* __restrict__ pWe2e,
            const float* __restrict__ bias_e, float* __restrict__ summed,
            float* __restrict__ deg, float* __restrict__ e_out, int E, int N) {
  __shared__ float ein_lds[16 * 64];
  __shared__ float e2_lds[16 * 64];
  __shared__ float h1s_lds[16 * 64];
  __shared__ int   src_lds[16];
  __shared__ int   dst_lds[16];

  const int e0 = blockIdx.x * 16;
  if (threadIdx.x < 16) {
    int e = min(e0 + (int)threadIdx.x, E - 1);
    src_lds[threadIdx.x] = src[e];
    dst_lds[threadIdx.x] = dst[e];
  }
  __syncthreads();

  for (int idx = threadIdx.x; idx < 16 * 64; idx += 128) {
    int r = idx >> 6, c = idx & 63;
    int e = min(e0 + r, E - 1);
    int s = src_lds[r], d = dst_lds[r];
    ein_lds[idx] = Ein[(size_t)e * 64 + c];
    e2_lds[idx]  = hu[(size_t)s * 64 + c] + hv[(size_t)d * 64 + c];
    h1s_lds[idx] = h1[(size_t)s * 64 + c];
  }
  __syncthreads();

  const int lane = threadIdx.x & 31;
  const int nt   = threadIdx.x >> 5;
  const int half = lane >> 4;
  const float* erow  = ein_lds + (lane & 15) * 64;
  const float* e2row = e2_lds  + (lane & 15) * 64;

  v8f acc1 = {};   // e1
  v8f acc2 = {};   // e_out (pre-bias)
#pragma unroll
  for (int kt = 0; kt < 2; ++kt) {
    v16bf a = pack_a(erow, kt * 32, half);
    acc1 = wmma_bf16(a, load_b(pWe2n, kt, nt, lane), acc1);
    acc2 = wmma_bf16(a, load_b(pWe2e, kt, nt, lane), acc2);
  }
#pragma unroll
  for (int kt = 2; kt < 4; ++kt) {                   // K 64..127 = e2
    v16bf a = pack_a(e2row, (kt - 2) * 32, half);
    acc2 = wmma_bf16(a, load_b(pWe2e, kt, nt, lane), acc2);
  }

  const int col = nt * 16 + (lane & 15);
  const int mb  = half * 8;
  const float be = bias_e[col];
#pragma unroll
  for (int r = 0; r < 8; ++r) {
    int mrow = mb + r;
    int e = e0 + mrow;
    if (e < E) {
      float mval = acc1[r] + h1s_lds[mrow * 64 + col];
      atomicAdd(&summed[(size_t)dst_lds[mrow] * 64 + col], mval);
      e_out[(size_t)e * 64 + col] = acc2[r] + be;
    }
  }
  if (nt == 0 && threadIdx.x < 16) {
    int e = e0 + (int)threadIdx.x;
    if (e < E) atomicAdd(&deg[dst_lds[threadIdx.x]], 1.0f);
  }
}

// ---------------------------------------------------------------------------
// Node output: h_out = [S, X, summed/max(deg,1)] * Wv + bias_n.
// ---------------------------------------------------------------------------
__global__ void __launch_bounds__(128)
node_out_kernel(const float* __restrict__ S, const float* __restrict__ X,
                const float* __restrict__ summed, const float* __restrict__ deg,
                const __bf16* __restrict__ pWv, const float* __restrict__ bias_n,
                float* __restrict__ h_out, int N) {
  __shared__ float s_lds[16 * 64];
  __shared__ float x_lds[16 * 64];
  __shared__ float hn_lds[16 * 64];
  const int row0 = blockIdx.x * 16;

  for (int idx = threadIdx.x; idx < 16 * 64; idx += 128) {
    int r = idx >> 6, c = idx & 63;
    int node = min(row0 + r, N - 1);
    s_lds[idx] = S[(size_t)node * 64 + c];
    x_lds[idx] = X[(size_t)node * 64 + c];
    float dg = deg[node];
    hn_lds[idx] = summed[(size_t)node * 64 + c] / fmaxf(dg, 1.0f);
  }
  __syncthreads();

  const int lane = threadIdx.x & 31;
  const int nt   = threadIdx.x >> 5;
  const int half = lane >> 4;
  const float* srow = s_lds  + (lane & 15) * 64;
  const float* xrow = x_lds  + (lane & 15) * 64;
  const float* hrow = hn_lds + (lane & 15) * 64;

  v8f acc = {};
#pragma unroll
  for (int kt = 0; kt < 2; ++kt)
    acc = wmma_bf16(pack_a(srow, kt * 32, half), load_b(pWv, kt, nt, lane), acc);
#pragma unroll
  for (int kt = 2; kt < 4; ++kt)
    acc = wmma_bf16(pack_a(xrow, (kt - 2) * 32, half), load_b(pWv, kt, nt, lane), acc);
#pragma unroll
  for (int kt = 4; kt < 6; ++kt)
    acc = wmma_bf16(pack_a(hrow, (kt - 4) * 32, half), load_b(pWv, kt, nt, lane), acc);

  const int col = nt * 16 + (lane & 15);
  const int mb  = half * 8;
  const float bn = bias_n[col];
#pragma unroll
  for (int r = 0; r < 8; ++r) {
    int node = row0 + mb + r;
    if (node < N) h_out[(size_t)node * 64 + col] = acc[r] + bn;
  }
}

// ---------------------------------------------------------------------------
extern "C" void kernel_launch(void* const* d_in, const int* in_sizes, int n_in,
                              void* d_out, int out_size, void* d_ws, size_t ws_size,
                              hipStream_t stream) {
  const float* S      = (const float*)d_in[0];
  const float* X      = (const float*)d_in[1];
  const float* Ein    = (const float*)d_in[2];
  const int*   src    = (const int*)d_in[3];
  const int*   dst    = (const int*)d_in[4];
  const float* W_u    = (const float*)d_in[5];   // 64x64
  const float* W_v    = (const float*)d_in[6];   // 192x64
  const float* W_e2n  = (const float*)d_in[7];   // 64x64
  const float* bias_n = (const float*)d_in[8];
  const float* W_eu   = (const float*)d_in[9];   // 64x64
  const float* W_ev   = (const float*)d_in[10];  // 128x64
  const float* W_e2e  = (const float*)d_in[11];  // 128x64
  const float* bias_e = (const float*)d_in[12];

  const int N = in_sizes[0] / 64;
  const int E = in_sizes[3];

  char* ws = (char*)d_ws;
  const size_t n64 = (size_t)N * 64 * sizeof(float);          // multiple of 1KB
  float* h1     = (float*)(ws);
  float* hu     = (float*)(ws + n64);
  float* hv     = (float*)(ws + 2 * n64);
  float* summed = (float*)(ws + 3 * n64);
  float* deg    = (float*)(ws + 4 * n64);
  const size_t degB = (((size_t)N * sizeof(float)) + 1023) & ~(size_t)1023;
  __bf16* packed = (__bf16*)(ws + 4 * n64 + degB);
  __bf16* pWu   = packed;          // K=64  -> 4096 bf16
  __bf16* pWeu  = pWu + 4096;      // K=64  -> 4096
  __bf16* pWe2n = pWeu + 4096;     // K=64  -> 4096
  __bf16* pWev  = pWe2n + 4096;    // K=128 -> 8192
  __bf16* pWe2e = pWev + 8192;     // K=128 -> 8192
  __bf16* pWv   = pWe2e + 8192;    // K=192 -> 12288

  // Zero segment-sum accumulators every call (deterministic; capture-safe).
  hipMemsetAsync(ws + 3 * n64, 0, n64 + (size_t)N * sizeof(float), stream);

  // Pack all weights to bf16 WMMA-B tile layout.
  pack_weight_kernel<<<(64 * 64 + 255) / 256, 256, 0, stream>>>(W_u, pWu, 64);
  pack_weight_kernel<<<(64 * 64 + 255) / 256, 256, 0, stream>>>(W_eu, pWeu, 64);
  pack_weight_kernel<<<(64 * 64 + 255) / 256, 256, 0, stream>>>(W_e2n, pWe2n, 64);
  pack_weight_kernel<<<(128 * 64 + 255) / 256, 256, 0, stream>>>(W_ev, pWev, 128);
  pack_weight_kernel<<<(128 * 64 + 255) / 256, 256, 0, stream>>>(W_e2e, pWe2e, 128);
  pack_weight_kernel<<<(192 * 64 + 255) / 256, 256, 0, stream>>>(W_v, pWv, 192);

  float* h_out = (float*)d_out;
  float* e_out = (float*)d_out + (size_t)N * 64;

  const int nodeTiles = (N + 15) / 16;
  const int edgeTiles = (E + 15) / 16;

  node_pre_kernel<<<nodeTiles, 128, 0, stream>>>(S, X, pWu, pWeu, pWev,
                                                 h1, hu, hv, N);
  edge_kernel<<<edgeTiles, 128, 0, stream>>>(Ein, src, dst, h1, hu, hv,
                                             pWe2n, pWe2e, bias_e,
                                             summed, deg, e_out, E, N);
  node_out_kernel<<<nodeTiles, 128, 0, stream>>>(S, X, summed, deg,
                                                 pWv, bias_n, h_out, N);
}
